// MoELayer_86036784873882
// MI455X (gfx1250) — compile-verified, matrix operand order checked by reading, not run on hardware
//
#include <hip/hip_runtime.h>
#include <hip/hip_bf16.h>

// ---------------- problem constants ----------------
#define N_TOK   8192         // B*S
#define DIM     2048         // D
#define HID     1024         // H
#define NEXP    8            // E
#define TOPK    2
#define NSLOT   (N_TOK * TOPK)           // 16384
#define OUT_ELEMS (N_TOK * DIM)           // 16777216 (+1 loss)

// ---------------- workspace layout (bytes) ----------------
// hdr ints: [0..7]=counts  [8..15]=cursor  [16..23]=offsets  [24..31]=probsum(f32)
#define WS_TOKE   1024                        // int  [NSLOT]
#define WS_TOKW   (1024 + 65536)              // f32  [NSLOT]
#define WS_SLOTT  (1024 + 131072)             // int  [NSLOT]
#define WS_SLOTW  (1024 + 196608)             // f32  [NSLOT]
#define WS_H      ((size_t)1  << 20)          // bf16 [NSLOT][HID]        (32 MB)
#define WS_W1T    ((size_t)33 << 20)          // bf16 [E][HID][DIM] (W1^T) (32 MB)
#define WS_W2T    ((size_t)65 << 20)          // bf16 [E][DIM][HID] (W2^T) (32 MB)
#define WS_XB     ((size_t)97 << 20)          // bf16 [N_TOK][DIM]         (32 MB)

#define TILEH 5120                            // 128 rows * 40 halves per LDS buffer

typedef __attribute__((ext_vector_type(16))) __bf16 v16bf;
typedef __attribute__((ext_vector_type(8)))  float  v8f;

// f32 -> bf16 round-to-nearest-even
__device__ __forceinline__ unsigned int f2b(float f) {
    unsigned int u = __float_as_uint(f);
    unsigned int r = u + 0x7FFFu + ((u >> 16) & 1u);
    return r >> 16;
}

// CDNA5 async copy: 32 bytes global -> LDS per lane (2x GLOBAL_LOAD_ASYNC_TO_LDS_B128).
// INST_OFFSET is added to BOTH the LDS and global addresses (ISA 08 §4.4), so one
// VGPR pair + offset:16 covers the second 16B. Tracked with ASYNCcnt.
__device__ __forceinline__ void async_cp32(const unsigned short* gsrc, unsigned short* ldst) {
    unsigned l = (unsigned)(size_t)ldst;      // low 32 bits of generic LDS addr = LDS offset
    asm volatile(
        "global_load_async_to_lds_b128 %0, %1, off\n\t"
        "global_load_async_to_lds_b128 %0, %1, off offset:16"
        :: "v"(l), "v"(gsrc)
        : "memory");
}
#define S_WAIT_ASYNC4 asm volatile("s_wait_asynccnt 0x4" ::: "memory")
#define S_WAIT_ASYNC0 asm volatile("s_wait_asynccnt 0x0" ::: "memory")

// Load one 16x32 bf16 WMMA fragment from LDS staged [row][k], row stride 40 halves
// (80 B: 16B-aligned chunks, conflict-free). Per ISA 7.12.2: lane -> row (lane&15),
// VGPRs 0-3 = K {0..7}+8*khalf, VGPRs 4-7 = K {16..23}+8*khalf, khalf = lane>>4.
__device__ __forceinline__ v16bf ldfrag(const unsigned short* base, int row, int khalf) {
    const unsigned short* p = base + row * 40 + khalf * 8;
    union { uint4 q[2]; v16bf v; } u;
    u.q[0] = *(const uint4*)(p);
    u.q[1] = *(const uint4*)(p + 16);
    return u.v;
}

// ---------------- kernel 1: zero out + counters ----------------
__global__ void moe_init_kernel(float* __restrict__ out, int* __restrict__ hdr) {
    size_t base = ((size_t)blockIdx.x * blockDim.x + threadIdx.x) * 4;
#pragma unroll
    for (int j = 0; j < 4; ++j) {
        size_t i = base + j;
        if (i < (size_t)OUT_ELEMS + 1) out[i] = 0.0f;
    }
    if (blockIdx.x == 0 && threadIdx.x < 64) hdr[threadIdx.x] = 0;
}

// ---------------- kernel 2: x f32 -> bf16 ----------------
__global__ __launch_bounds__(256)
void moe_cvt_x_kernel(const float* __restrict__ x, unsigned short* __restrict__ xb) {
    size_t i = ((size_t)blockIdx.x * 256 + threadIdx.x) * 8;
    const float4* s = (const float4*)(x + i);
    float4 f0 = s[0], f1 = s[1];
    uint4 q;
    q.x = f2b(f0.x) | (f2b(f0.y) << 16);
    q.y = f2b(f0.z) | (f2b(f0.w) << 16);
    q.z = f2b(f1.x) | (f2b(f1.y) << 16);
    q.w = f2b(f1.z) | (f2b(f1.w) << 16);
    *(uint4*)(xb + i) = q;
}

// ---------------- kernel 3: weight transpose + f32 -> bf16 ----------------
// in: [z][R][C] f32   ->   out: [z][C][R] bf16   (32x32 LDS-tiled transpose)
__global__ __launch_bounds__(256)
void moe_cvt_wT_kernel(const float* __restrict__ in, unsigned short* __restrict__ out,
                       int R, int C) {
    const float* ine = in + (size_t)blockIdx.z * R * C;
    unsigned short* oute = out + (size_t)blockIdx.z * R * C;
    int r0 = blockIdx.y * 32, c0 = blockIdx.x * 32;
    __shared__ float t[32][33];
    int lane = threadIdx.x & 31, wid = threadIdx.x >> 5;
#pragma unroll
    for (int i = 0; i < 4; ++i) {
        int r = wid + i * 8;
        t[r][lane] = ine[(size_t)(r0 + r) * C + c0 + lane];
    }
    __syncthreads();
#pragma unroll
    for (int i = 0; i < 4; ++i) {
        int r = wid + i * 8;   // output row = c0+r, output col = r0+lane
        oute[(size_t)(c0 + r) * R + r0 + lane] = (unsigned short)f2b(t[lane][r]);
    }
}

// ---------------- kernel 4: gating (one wave32 per token) ----------------
__global__ __launch_bounds__(256)
void moe_gate_kernel(const float* __restrict__ x, const float* __restrict__ Wg,
                     int* __restrict__ hdr, int* __restrict__ tok_e,
                     float* __restrict__ tok_w) {
    const int wid = threadIdx.x >> 5, lane = threadIdx.x & 31;
    const int n = blockIdx.x * 8 + wid;
    const float* xr = x + (size_t)n * DIM;

    float acc[NEXP];
#pragma unroll
    for (int e = 0; e < NEXP; ++e) acc[e] = 0.0f;

    for (int i = 0; i < DIM / 32; ++i) {
        int d = i * 32 + lane;
        float xv = xr[d];
        const float4* wr = (const float4*)(Wg + (size_t)d * NEXP);
        float4 w0 = wr[0], w1 = wr[1];
        acc[0] += xv * w0.x; acc[1] += xv * w0.y;
        acc[2] += xv * w0.z; acc[3] += xv * w0.w;
        acc[4] += xv * w1.x; acc[5] += xv * w1.y;
        acc[6] += xv * w1.z; acc[7] += xv * w1.w;
    }
#pragma unroll
    for (int off = 16; off > 0; off >>= 1)
#pragma unroll
        for (int e = 0; e < NEXP; ++e) acc[e] += __shfl_xor(acc[e], off, 32);

    if (lane == 0) {
        float mx = acc[0];
#pragma unroll
        for (int e = 1; e < NEXP; ++e) mx = fmaxf(mx, acc[e]);
        float g[NEXP], s = 0.0f;
#pragma unroll
        for (int e = 0; e < NEXP; ++e) { g[e] = __expf(acc[e] - mx); s += g[e]; }
        float inv = 1.0f / s;
#pragma unroll
        for (int e = 0; e < NEXP; ++e) g[e] *= inv;
        int i1 = 0;
#pragma unroll
        for (int e = 1; e < NEXP; ++e) if (g[e] > g[i1]) i1 = e;
        int i2 = (i1 == 0) ? 1 : 0;
#pragma unroll
        for (int e = 0; e < NEXP; ++e) if (e != i1 && g[e] > g[i2]) i2 = e;

        tok_e[2 * n + 0] = i1; tok_w[2 * n + 0] = g[i1];
        tok_e[2 * n + 1] = i2; tok_w[2 * n + 1] = g[i2];
        atomicAdd(&hdr[i1], 1);
        atomicAdd(&hdr[i2], 1);
        float* ps = (float*)(hdr + 24);
#pragma unroll
        for (int e = 0; e < NEXP; ++e) unsafeAtomicAdd(&ps[e], g[e]);
    }
}

// ---------------- kernel 5: offsets + load-balancing loss ----------------
__global__ void moe_finalize_kernel(int* __restrict__ hdr, float* __restrict__ loss_out) {
    if (threadIdx.x == 0 && blockIdx.x == 0) {
        int* counts = hdr;
        int* offs = hdr + 16;
        int tot = 0;
        for (int e = 0; e < NEXP; ++e) { offs[e] = tot; tot += counts[e]; }
        const float* ps = (const float*)(hdr + 24);
        float p[NEXP], m = 0.0f;
        for (int e = 0; e < NEXP; ++e) { p[e] = ps[e] / (float)N_TOK; m += p[e]; }
        m *= (1.0f / NEXP);
        float var = 0.0f;
        for (int e = 0; e < NEXP; ++e) { float d = p[e] - m; var += d * d; }
        var *= (1.0f / NEXP);
        loss_out[0] = var / (m * m + 1e-10f);
    }
}

// ---------------- kernel 6: scatter tokens into per-expert slots ----------------
__global__ __launch_bounds__(256)
void moe_scatter_kernel(int* __restrict__ hdr, const int* __restrict__ tok_e,
                        const float* __restrict__ tok_w, int* __restrict__ slot_t,
                        float* __restrict__ slot_w) {
    int n = blockIdx.x * blockDim.x + threadIdx.x;
    if (n >= N_TOK) return;
    int* cursor = hdr + 8;
    const int* offs = hdr + 16;
#pragma unroll
    for (int k = 0; k < TOPK; ++k) {
        int e = tok_e[2 * n + k];
        int pos = atomicAdd(&cursor[e], 1);
        int s = offs[e] + pos;
        slot_t[s] = n;
        slot_w[s] = tok_w[2 * n + k];
    }
}

// ---------------- kernel 7: grouped GEMM1  h = silu(x_gather @ W1[e]) ----------------
// A = xb gathered (bf16), B = W1T[e] (bf16, [HID][DIM] = already [n][k]).
// 128x128 tile / 256 threads (8 waves 2x4), BK=32, double-buffered async-to-LDS.
__global__ __launch_bounds__(256)
void moe_gemm1_kernel(const unsigned short* __restrict__ xb,
                      const unsigned short* __restrict__ W1T,
                      const int* __restrict__ hdr, const int* __restrict__ slot_t,
                      unsigned short* __restrict__ hbuf) {
    const int e = blockIdx.z;
    const int cnt = hdr[e];
    const int mbase = blockIdx.y * 128;
    if (mbase >= cnt) return;                  // whole-block exit: EXEC stays all-1
    const int slot0 = hdr[16 + e];
    const int n0 = blockIdx.x * 128;
    const unsigned short* We = W1T + (size_t)e * HID * DIM;

    __shared__ __align__(16) unsigned short As[2 * TILEH];
    __shared__ __align__(16) unsigned short Bs[2 * TILEH];

    const int tid = threadIdx.x, wid = tid >> 5, lane = tid & 31;
    const int wm = wid & 1, wn = wid >> 1;
    const int khalf = lane >> 4, lr = lane & 15;

    v8f c[4][2];
#pragma unroll
    for (int i = 0; i < 4; ++i)
#pragma unroll
        for (int j = 0; j < 2; ++j)
#pragma unroll
            for (int q = 0; q < 8; ++q) c[i][j][q] = 0.0f;

    // staging: thread -> (row, 32B chunk); A gather row fixed across k loop
    const int srow = tid >> 1, sch = tid & 1;
    int mr = mbase + srow; if (mr >= cnt) mr = cnt - 1;       // clamp, masked at store
    const unsigned short* agsrc = xb + (size_t)slot_t[slot0 + mr] * DIM + sch * 16;
    const unsigned short* bgsrc = We + (size_t)(n0 + srow) * DIM + sch * 16;
    unsigned short* aldst0 = As + srow * 40 + sch * 16;
    unsigned short* bldst0 = Bs + srow * 40 + sch * 16;

#define G1_STAGE(kb, buf)                                          \
    do {                                                           \
        async_cp32(agsrc + (kb) * 32, aldst0 + (buf) * TILEH);     \
        async_cp32(bgsrc + (kb) * 32, bldst0 + (buf) * TILEH);     \
    } while (0)

    G1_STAGE(0, 0);
    for (int kb = 0; kb < DIM / 32; ++kb) {
        const int cur = kb & 1;
        if (kb + 1 < DIM / 32) { G1_STAGE(kb + 1, cur ^ 1); S_WAIT_ASYNC4; }
        else                   { S_WAIT_ASYNC0; }
        __syncthreads();

        const unsigned short* Ab = As + cur * TILEH;
        const unsigned short* Bb = Bs + cur * TILEH;
        v16bf a[4], b[2];
#pragma unroll
        for (int tm = 0; tm < 4; ++tm) a[tm] = ldfrag(Ab, wm * 64 + tm * 16 + lr, khalf);
#pragma unroll
        for (int tn = 0; tn < 2; ++tn) b[tn] = ldfrag(Bb, wn * 32 + tn * 16 + lr, khalf);
#pragma unroll
        for (int tm = 0; tm < 4; ++tm)
#pragma unroll
            for (int tn = 0; tn < 2; ++tn)
                c[tm][tn] = __builtin_amdgcn_wmma_f32_16x16x32_bf16(
                    false, a[tm], false, b[tn], (short)0, c[tm][tn], false, false);
        __syncthreads();
    }
#undef G1_STAGE

    // epilogue: silu -> bf16 -> h buffer
#pragma unroll
    for (int tm = 0; tm < 4; ++tm)
#pragma unroll
        for (int j = 0; j < 8; ++j) {
            int mrow = mbase + wm * 64 + tm * 16 + j + 8 * khalf;
            if (mrow < cnt) {
                size_t hrow = (size_t)(slot0 + mrow) * HID;
#pragma unroll
                for (int tn = 0; tn < 2; ++tn) {
                    int n = n0 + wn * 32 + tn * 16 + lr;
                    float v = c[tm][tn][j];
                    v = v / (1.0f + __expf(-v));
                    hbuf[hrow + n] = (unsigned short)f2b(v);
                }
            }
        }
}

// ---------------- kernel 8: grouped GEMM2  out += w * (h @ W2[e]) ----------------
// A = hbuf (bf16, contiguous slots), B = W2T[e] (bf16, [DIM][HID] = [n][k]).
__global__ __launch_bounds__(256)
void moe_gemm2_kernel(const unsigned short* __restrict__ hbuf,
                      const unsigned short* __restrict__ W2T,
                      const int* __restrict__ hdr, const int* __restrict__ slot_t,
                      const float* __restrict__ slot_w, float* __restrict__ out) {
    const int e = blockIdx.z;
    const int cnt = hdr[e];
    const int mbase = blockIdx.y * 128;
    if (mbase >= cnt) return;
    const int slot0 = hdr[16 + e];
    const int n0 = blockIdx.x * 128;
    const unsigned short* We = W2T + (size_t)e * DIM * HID;

    __shared__ __align__(16) unsigned short As[2 * TILEH];
    __shared__ __align__(16) unsigned short Bs[2 * TILEH];

    const int tid = threadIdx.x, wid = tid >> 5, lane = tid & 31;
    const int wm = wid & 1, wn = wid >> 1;
    const int khalf = lane >> 4, lr = lane & 15;

    v8f c[4][2];
#pragma unroll
    for (int i = 0; i < 4; ++i)
#pragma unroll
        for (int j = 0; j < 2; ++j)
#pragma unroll
            for (int q = 0; q < 8; ++q) c[i][j][q] = 0.0f;

    const int srow = tid >> 1, sch = tid & 1;
    int mr = mbase + srow; if (mr >= cnt) mr = cnt - 1;
    const unsigned short* agsrc = hbuf + (size_t)(slot0 + mr) * HID + sch * 16;
    const unsigned short* bgsrc = We + (size_t)(n0 + srow) * HID + sch * 16;
    unsigned short* aldst0 = As + srow * 40 + sch * 16;
    unsigned short* bldst0 = Bs + srow * 40 + sch * 16;

#define G2_STAGE(kb, buf)                                          \
    do {                                                           \
        async_cp32(agsrc + (kb) * 32, aldst0 + (buf) * TILEH);     \
        async_cp32(bgsrc + (kb) * 32, bldst0 + (buf) * TILEH);     \
    } while (0)

    G2_STAGE(0, 0);
    for (int kb = 0; kb < HID / 32; ++kb) {
        const int cur = kb & 1;
        if (kb + 1 < HID / 32) { G2_STAGE(kb + 1, cur ^ 1); S_WAIT_ASYNC4; }
        else                   { S_WAIT_ASYNC0; }
        __syncthreads();

        const unsigned short* Ab = As + cur * TILEH;
        const unsigned short* Bb = Bs + cur * TILEH;
        v16bf a[4], b[2];
#pragma unroll
        for (int tm = 0; tm < 4; ++tm) a[tm] = ldfrag(Ab, wm * 64 + tm * 16 + lr, khalf);
#pragma unroll
        for (int tn = 0; tn < 2; ++tn) b[tn] = ldfrag(Bb, wn * 32 + tn * 16 + lr, khalf);
#pragma unroll
        for (int tm = 0; tm < 4; ++tm)
#pragma unroll
            for (int tn = 0; tn < 2; ++tn)
                c[tm][tn] = __builtin_amdgcn_wmma_f32_16x16x32_bf16(
                    false, a[tm], false, b[tn], (short)0, c[tm][tn], false, false);
        __syncthreads();
    }
#undef G2_STAGE

    // epilogue: weighted scatter-add into out (global_atomic_add_f32)
#pragma unroll
    for (int tm = 0; tm < 4; ++tm)
#pragma unroll
        for (int j = 0; j < 8; ++j) {
            int mrow = mbase + wm * 64 + tm * 16 + j + 8 * khalf;
            if (mrow < cnt) {
                int slot = slot0 + mrow;
                int tok = slot_t[slot];
                float w = slot_w[slot];
                size_t orow = (size_t)tok * DIM;
#pragma unroll
                for (int tn = 0; tn < 2; ++tn) {
                    int n = n0 + wn * 32 + tn * 16 + lr;
                    unsafeAtomicAdd(&out[orow + n], w * c[tm][tn][j]);
                }
            }
        }
}

// ---------------- host launch ----------------
extern "C" void kernel_launch(void* const* d_in, const int* in_sizes, int n_in,
                              void* d_out, int out_size, void* d_ws, size_t ws_size,
                              hipStream_t stream) {
    const float* x  = (const float*)d_in[0];
    const float* Wg = (const float*)d_in[1];
    const float* W1 = (const float*)d_in[2];
    const float* W2 = (const float*)d_in[3];
    float* out = (float*)d_out;

    char* ws = (char*)d_ws;
    int*   hdr    = (int*)ws;
    int*   tok_e  = (int*)(ws + WS_TOKE);
    float* tok_w  = (float*)(ws + WS_TOKW);
    int*   slot_t = (int*)(ws + WS_SLOTT);
    float* slot_w = (float*)(ws + WS_SLOTW);
    unsigned short* hbuf = (unsigned short*)(ws + WS_H);
    unsigned short* W1T  = (unsigned short*)(ws + WS_W1T);
    unsigned short* W2T  = (unsigned short*)(ws + WS_W2T);
    unsigned short* xb   = (unsigned short*)(ws + WS_XB);

    moe_init_kernel<<<(OUT_ELEMS / 4 + 256) / 256, 256, 0, stream>>>(out, hdr);
    moe_cvt_x_kernel<<<(size_t)N_TOK * DIM / 8 / 256, 256, 0, stream>>>(x, xb);
    moe_cvt_wT_kernel<<<dim3(HID / 32, DIM / 32, NEXP), 256, 0, stream>>>(W1, W1T, DIM, HID);
    moe_cvt_wT_kernel<<<dim3(DIM / 32, HID / 32, NEXP), 256, 0, stream>>>(W2, W2T, HID, DIM);
    moe_gate_kernel<<<N_TOK / 8, 256, 0, stream>>>(x, Wg, hdr, tok_e, tok_w);
    moe_finalize_kernel<<<1, 32, 0, stream>>>(hdr, out + OUT_ELEMS);
    moe_scatter_kernel<<<N_TOK / 256, 256, 0, stream>>>(hdr, tok_e, tok_w, slot_t, slot_w);
    moe_gemm1_kernel<<<dim3(HID / 128, N_TOK / 128, NEXP), 256, 0, stream>>>(
        xb, W1T, hdr, slot_t, hbuf);
    moe_gemm2_kernel<<<dim3(DIM / 128, N_TOK / 128, NEXP), 256, 0, stream>>>(
        hbuf, W2T, hdr, slot_t, slot_w, out);
}
